// _Structure_inference_10780367913387
// MI455X (gfx1250) — matrix operand order repaired and checked by reading, not compiled, not run on hardware
//
#include <hip/hip_runtime.h>
#include <cmath>

#define NB   8192
#define DD   2048
#define GDIM (3 * DD)

typedef __attribute__((ext_vector_type(16))) __bf16 v16bf;
typedef __attribute__((ext_vector_type(8)))  __bf16 v8bf;
typedef __attribute__((ext_vector_type(4)))  __bf16 v4bf;
typedef __attribute__((ext_vector_type(8)))  float  v8f;

__device__ __forceinline__ v16bf cat8(v8bf lo, v8bf hi) {
  return __builtin_shufflevector(lo, hi, 0,1,2,3,4,5,6,7,8,9,10,11,12,13,14,15);
}

// B operand: 16 contiguous bf16 (one column's K-run) -> v16bf
__device__ __forceinline__ v16bf loadB(const __bf16* __restrict__ p) {
  v8bf lo = *(const v8bf*)(p);
  v8bf hi = *(const v8bf*)(p + 8);
  return cat8(lo, hi);
}

// A operand from LDS row: elements 0..15 contiguous at +0
__device__ __forceinline__ v16bf loadA(const __bf16* p) {
  v8bf lo = *(const v8bf*)(p);
  v8bf hi = *(const v8bf*)(p + 8);
  return cat8(lo, hi);
}

__device__ __forceinline__ float sigm(float x) { return 1.0f / (1.0f + __expf(-x)); }

// ---------------- weight fp32 -> bf16 ----------------
__global__ void k_convert_w(const float* __restrict__ w, __bf16* __restrict__ o, int n4) {
  int i = (blockIdx.x * blockDim.x + threadIdx.x) * 4;
  if (i < n4 * 4) {
    float4 v = *(const float4*)(w + i);
    v4bf b;
    b[0] = (__bf16)v.x; b[1] = (__bf16)v.y; b[2] = (__bf16)v.z; b[3] = (__bf16)v.w;
    *(v4bf*)(o + i) = b;
  }
}

// ---------------- activations fp32 -> bf16: hb = bf16(hi), xb = bf16(z[row]*hi) ----------------
__global__ void k_prep_act(const float* __restrict__ hi_in, const float* __restrict__ z,
                           __bf16* __restrict__ hb, __bf16* __restrict__ xb) {
  int i = (blockIdx.x * blockDim.x + threadIdx.x) * 4;
  float zr = z[i >> 11];  // row = i / DD
  float4 v = *(const float4*)(hi_in + i);
  v4bf h, x;
  h[0] = (__bf16)v.x;        h[1] = (__bf16)v.y;
  h[2] = (__bf16)v.z;        h[3] = (__bf16)v.w;
  x[0] = (__bf16)(zr * v.x); x[1] = (__bf16)(zr * v.y);
  x[2] = (__bf16)(zr * v.z); x[3] = (__bf16)(zr * v.w);
  *(v4bf*)(hb + i) = h;
  *(v4bf*)(xb + i) = x;
}

// ---------------- scalars: PA0, s1_0 = hi[0].w1, s2c = whole.w2 ----------------
__global__ void k_scal(const float* __restrict__ edges, const float* __restrict__ wua,
                       const float* __restrict__ hi_in, const float* __restrict__ cw2,
                       const float* __restrict__ whole, float* __restrict__ scal) {
  __shared__ float r1[256], r2[256];
  int t = threadIdx.x;
  float s1 = 0.f, s2 = 0.f;
  for (int k = t; k < DD; k += 256) {
    s1 += hi_in[k] * cw2[k];
    s2 += whole[k] * cw2[DD + k];
  }
  r1[t] = s1; r2[t] = s2; __syncthreads();
  for (int s = 128; s > 0; s >>= 1) {
    if (t < s) { r1[t] += r1[t + s]; r2[t] += r2[t + s]; }
    __syncthreads();
  }
  if (t == 0) {
    float pa = 0.f;
    for (int i = 0; i < 9; ++i) pa += edges[i] * wua[i];
    scal[0] = fmaxf(pa, 0.f);
    scal[1] = r1[0];
    scal[2] = r2[0];
  }
}

// ---------------- length-N softmax of row 0 of attention ----------------
__global__ void k_softmax(const float* __restrict__ scal, const float* __restrict__ d2,
                          float* __restrict__ atmp, float* __restrict__ z, int step) {
  __shared__ float red[1024];
  int t = threadIdx.x;
  float pa = scal[0], s1 = scal[1], s2c = scal[2];
  float mx = -1e30f;
  for (int j = t; j < NB; j += 1024) {
    float s2 = (step == 0) ? s2c : z[j] * d2[j];
    float a = pa * tanhf(s1 + s2);
    atmp[j] = a;
    mx = fmaxf(mx, a);
  }
  red[t] = mx; __syncthreads();
  for (int s = 512; s > 0; s >>= 1) { if (t < s) red[t] = fmaxf(red[t], red[t + s]); __syncthreads(); }
  mx = red[0]; __syncthreads();
  float sm = 0.f;
  for (int j = t; j < NB; j += 1024) sm += __expf(atmp[j] - mx);
  red[t] = sm; __syncthreads();
  for (int s = 512; s > 0; s >>= 1) { if (t < s) red[t] += red[t + s]; __syncthreads(); }
  float inv = 1.0f / red[0];
  __syncthreads();
  for (int j = t; j < NB; j += 1024) z[j] = __expf(atmp[j] - mx) * inv;
}

// ---------------- d2[j] = hi[j] . w2 (for the next step's s2) ----------------
__global__ void k_d2(const float* __restrict__ hi_in, const float* __restrict__ cw2,
                     float* __restrict__ d2) {
  int wave = threadIdx.x >> 5, lane = threadIdx.x & 31;
  int row = blockIdx.x * 8 + wave;
  const float* hr = hi_in + (size_t)row * DD;
  const float* w2 = cw2 + DD;
  float s = 0.f;
  for (int k = lane * 4; k < DD; k += 128) {
    float4 a = *(const float4*)(hr + k);
    float4 b = *(const float4*)(w2 + k);
    s += a.x * b.x + a.y * b.y + a.z * b.z + a.w * b.w;
  }
  for (int off = 16; off; off >>= 1) s += __shfl_xor(s, off, 32);
  if (lane == 0) d2[row] = s;
}

// ---------------- fused GRU step: async LDS staging, M=32 tile, 12 WMMA/substep ----------------
__global__ void __launch_bounds__(256)
k_gru(const float* __restrict__ hi_in,
      const __bf16* __restrict__ hb, const __bf16* __restrict__ xb,
      const __bf16* __restrict__ wih, const __bf16* __restrict__ whh,
      const float* __restrict__ b_ih, const float* __restrict__ b_hh,
      const float* __restrict__ wconf, float* __restrict__ hi_out) {
  // [buffer][matrix: 0=Ah(hi), 1=Ax(z*hi)][32 rows x 64 K]
  __shared__ __bf16 lds[2][2][32 * 64];

  const int m0   = blockIdx.x * 32;
  const int c0   = blockIdx.y * 128;
  const int tid  = threadIdx.x;
  const int lane = tid & 31;
  const int wave = tid >> 5;
  const int ncol = lane & 15;
  const int hi16 = lane >> 4;        // 0 or 1
  const int cw   = c0 + wave * 16;   // this wave's 16-column slab

  v8f a_ir0 = {0,0,0,0,0,0,0,0}, a_iz0 = {0,0,0,0,0,0,0,0}, a_in0 = {0,0,0,0,0,0,0,0};
  v8f a_hr0 = {0,0,0,0,0,0,0,0}, a_hz0 = {0,0,0,0,0,0,0,0}, a_hn0 = {0,0,0,0,0,0,0,0};
  v8f a_ir1 = {0,0,0,0,0,0,0,0}, a_iz1 = {0,0,0,0,0,0,0,0}, a_in1 = {0,0,0,0,0,0,0,0};
  v8f a_hr1 = {0,0,0,0,0,0,0,0}, a_hz1 = {0,0,0,0,0,0,0,0}, a_hn1 = {0,0,0,0,0,0,0,0};

  // async staging: each thread DMAs 16B of Ah and 16B of Ax per buffer
  const int srow = tid >> 3;         // 0..31
  const int sk   = (tid & 7) * 8;    // 0..56
  const __bf16* gh = hb + (size_t)(m0 + srow) * DD + sk;
  const __bf16* gx = xb + (size_t)(m0 + srow) * DD + sk;
  const uint32_t ldsb = (uint32_t)(uintptr_t)(&lds[0][0][0]) +
                        (uint32_t)((srow * 64 + sk) * 2);

#define ISSUE_ASYNC(buf, kkv)                                                   \
  do {                                                                          \
    uint64_t gah_ = (uint64_t)(uintptr_t)(gh + (kkv));                          \
    uint64_t gax_ = (uint64_t)(uintptr_t)(gx + (kkv));                          \
    uint32_t lah_ = ldsb + (uint32_t)((buf) * (2 * 32 * 64 * 2));               \
    uint32_t lax_ = lah_ + (uint32_t)(32 * 64 * 2);                             \
    asm volatile("global_load_async_to_lds_b128 %0, %1, off"                    \
                 :: "v"(lah_), "v"(gah_) : "memory");                           \
    asm volatile("global_load_async_to_lds_b128 %0, %1, off"                    \
                 :: "v"(lax_), "v"(gax_) : "memory");                           \
  } while (0)

  // per-lane B base pointers: column (cw+ncol) of each gate's weight rows,
  // with the lane-group K offset (+16 elements for lanes 16..31)
  const size_t roff = (size_t)(cw + ncol) * DD + (hi16 ? 16 : 0);
  const __bf16* pir = wih + roff;
  const __bf16* piz = wih + (size_t)DD * DD + roff;
  const __bf16* pin = wih + (size_t)2 * DD * DD + roff;
  const __bf16* phr = whh + roff;
  const __bf16* phz = whh + (size_t)DD * DD + roff;
  const __bf16* phn = whh + (size_t)2 * DD * DD + roff;

  ISSUE_ASYNC(0, 0);

  for (int kk = 0; kk < DD; kk += 64) {
    const int cur = (kk >> 6) & 1;
    if (kk + 64 < DD) {
      ISSUE_ASYNC(cur ^ 1, kk + 64);                       // prefetch next tile pair
      asm volatile("s_wait_asynccnt 0x2" ::: "memory");    // in-order: current pair done
    } else {
      asm volatile("s_wait_asynccnt 0x0" ::: "memory");
    }
    __syncthreads();

#pragma unroll
    for (int ks = 0; ks < 64; ks += 32) {
      const int oa = (lane & 15) * 64 + ks + (hi16 ? 16 : 0);  // M-subtile 0 (rows 0..15)
      const int ob = oa + 16 * 64;                              // M-subtile 1 (rows 16..31)
      v16bf Ah0 = loadA(&lds[cur][0][oa]);
      v16bf Ah1 = loadA(&lds[cur][0][ob]);
      v16bf Ax0 = loadA(&lds[cur][1][oa]);
      v16bf Ax1 = loadA(&lds[cur][1][ob]);

      const int kb = kk + ks;
      v16bf Bir = loadB(pir + kb);
      v16bf Biz = loadB(piz + kb);
      v16bf Bin = loadB(pin + kb);
      v16bf Bhr = loadB(phr + kb);
      v16bf Bhz = loadB(phz + kb);
      v16bf Bhn = loadB(phn + kb);

      a_ir0 = __builtin_amdgcn_wmma_f32_16x16x32_bf16(false, Ax0, false, Bir, (short)0, a_ir0, false, false);
      a_ir1 = __builtin_amdgcn_wmma_f32_16x16x32_bf16(false, Ax1, false, Bir, (short)0, a_ir1, false, false);
      a_iz0 = __builtin_amdgcn_wmma_f32_16x16x32_bf16(false, Ax0, false, Biz, (short)0, a_iz0, false, false);
      a_iz1 = __builtin_amdgcn_wmma_f32_16x16x32_bf16(false, Ax1, false, Biz, (short)0, a_iz1, false, false);
      a_in0 = __builtin_amdgcn_wmma_f32_16x16x32_bf16(false, Ax0, false, Bin, (short)0, a_in0, false, false);
      a_in1 = __builtin_amdgcn_wmma_f32_16x16x32_bf16(false, Ax1, false, Bin, (short)0, a_in1, false, false);
      a_hr0 = __builtin_amdgcn_wmma_f32_16x16x32_bf16(false, Ah0, false, Bhr, (short)0, a_hr0, false, false);
      a_hr1 = __builtin_amdgcn_wmma_f32_16x16x32_bf16(false, Ah1, false, Bhr, (short)0, a_hr1, false, false);
      a_hz0 = __builtin_amdgcn_wmma_f32_16x16x32_bf16(false, Ah0, false, Bhz, (short)0, a_hz0, false, false);
      a_hz1 = __builtin_amdgcn_wmma_f32_16x16x32_bf16(false, Ah1, false, Bhz, (short)0, a_hz1, false, false);
      a_hn0 = __builtin_amdgcn_wmma_f32_16x16x32_bf16(false, Ah0, false, Bhn, (short)0, a_hn0, false, false);
      a_hn1 = __builtin_amdgcn_wmma_f32_16x16x32_bf16(false, Ah1, false, Bhn, (short)0, a_hn1, false, false);
    }
    __syncthreads();
  }

  // ---- fused GRU gate epilogue (bias depends only on this lane's column) ----
  const int colg = cw + ncol;
  const float bir = b_ih[colg],          bhr = b_hh[colg];
  const float biz = b_ih[colg + DD],     bhz = b_hh[colg + DD];
  const float bin = b_ih[colg + 2 * DD], bhn = b_hh[colg + 2 * DD];
  const float wc  = wconf[0];

#pragma unroll
  for (int e = 0; e < 8; ++e) {
    {  // M-subtile 0
      const int row = m0 + e + (hi16 ? 8 : 0);
      const size_t idx = (size_t)row * DD + colg;
      const float h  = hi_in[idx];
      const float r  = sigm(a_ir0[e] + bir + a_hr0[e] + bhr);
      const float zg = sigm(a_iz0[e] + biz + a_hz0[e] + bhz);
      const float ng = tanhf(a_in0[e] + bin + r * (a_hn0[e] + bhn));
      const float h1 = (1.0f - zg) * ng + zg * h;
      hi_out[idx] = fmaxf(h1 * wc + h, 0.0f);
    }
    {  // M-subtile 1
      const int row = m0 + 16 + e + (hi16 ? 8 : 0);
      const size_t idx = (size_t)row * DD + colg;
      const float h  = hi_in[idx];
      const float r  = sigm(a_ir1[e] + bir + a_hr1[e] + bhr);
      const float zg = sigm(a_iz1[e] + biz + a_hz1[e] + bhz);
      const float ng = tanhf(a_in1[e] + bin + r * (a_hn1[e] + bhn));
      const float h1 = (1.0f - zg) * ng + zg * h;
      hi_out[idx] = fmaxf(h1 * wc + h, 0.0f);
    }
  }
#undef ISSUE_ASYNC
}

extern "C" void kernel_launch(void* const* d_in, const int* in_sizes, int n_in,
                              void* d_out, int out_size, void* d_ws, size_t ws_size,
                              hipStream_t stream) {
  const float* edges = (const float*)d_in[0];
  const float* rois  = (const float*)d_in[1];
  const float* whole = (const float*)d_in[2];
  const float* wua   = (const float*)d_in[3];
  const float* cw2   = (const float*)d_in[4];
  const float* wconf = (const float*)d_in[5];
  const float* wih   = (const float*)d_in[6];
  const float* whh   = (const float*)d_in[7];
  const float* bih   = (const float*)d_in[8];
  const float* bhh   = (const float*)d_in[9];

  char* p = (char*)d_ws;
  __bf16* wihb = (__bf16*)p; p += (size_t)GDIM * DD * sizeof(__bf16);
  __bf16* whhb = (__bf16*)p; p += (size_t)GDIM * DD * sizeof(__bf16);
  __bf16* hbf  = (__bf16*)p; p += (size_t)NB * DD * sizeof(__bf16);
  __bf16* xbf  = (__bf16*)p; p += (size_t)NB * DD * sizeof(__bf16);
  float* hiB   = (float*)p;  p += (size_t)NB * DD * sizeof(float);
  float* d2    = (float*)p;  p += NB * sizeof(float);
  float* atmp  = (float*)p;  p += NB * sizeof(float);
  float* zb    = (float*)p;  p += NB * sizeof(float);
  float* scal  = (float*)p;  p += 64;

  const int n4 = (GDIM * DD) / 4;  // threads for weight conversion (4 elems each)
  k_convert_w<<<(n4 + 255) / 256, 256, 0, stream>>>(wih, wihb, n4);
  k_convert_w<<<(n4 + 255) / 256, 256, 0, stream>>>(whh, whhb, n4);

  const float* hin = rois;
  float* hout = hiB;
  for (int step = 0; step < 2; ++step) {
    k_scal<<<1, 256, 0, stream>>>(edges, wua, hin, cw2, whole, scal);
    k_softmax<<<1, 1024, 0, stream>>>(scal, d2, atmp, zb, step);
    k_prep_act<<<(NB * DD / 4) / 256, 256, 0, stream>>>(hin, zb, hbf, xbf);
    k_d2<<<NB / 8, 256, 0, stream>>>(hin, cw2, d2);
    dim3 grid(NB / 32, DD / 128);
    k_gru<<<grid, 256, 0, stream>>>(hin, hbf, xbf, wihb, whhb, bih, bhh, wconf, hout);
    hin = hout;
    hout = (float*)d_out;
  }
}
